// MultiAgentActorCritic_23819888624097
// MI455X (gfx1250) — matrix operand order repaired and checked by reading, not compiled, not run on hardware
//
#include <hip/hip_runtime.h>

typedef unsigned short u16;
typedef unsigned int   u32;
typedef __attribute__((ext_vector_type(16))) _Float16 v16h;
typedef __attribute__((ext_vector_type(8)))  float    v8f;
typedef __attribute__((ext_vector_type(4)))  u32      u32x4;

// ---- problem constants ----
constexpr int kB   = 1024;
constexpr int kN   = 16;
constexpr int kBN  = kB * kN;        // 16384
constexpr int kOBS = 128;
constexpr int kACT = 32;
constexpr int kRH  = 256;
constexpr int kSH  = 256;
constexpr int kK   = 8;
constexpr int kE   = 240;            // N*(N-1)
constexpr int kXK  = 160;            // OBS+1=129 padded to %32
constexpr int kPROJ = 1024;          // 4 concatenated 256-wide node projections
constexpr int kKOBS = 1024;          // K*OBS
constexpr int kDECN = 144;           // OBS+1=129 padded to %16

// output layout (floats, concatenated in return order)
constexpr size_t OFF_ACTOR = 0;
constexpr size_t OFF_CRI   = OFF_ACTOR + (size_t)kB * kN * kACT;   // 524288
constexpr size_t OFF_SCM   = OFF_CRI   + (size_t)kB * kN;          // 540672
constexpr size_t OFF_REL   = OFF_SCM   + (size_t)kB * kN * 129;    // 2654208
constexpr size_t OFF_H     = OFF_REL   + (size_t)kB * kE * kK;     // 4620288
constexpr int    SCM_ELEMS = kB * kN * 129;

// ---- f16 helpers ----
__device__ inline u16  f2h(float v) { _Float16 h = (_Float16)v; return __builtin_bit_cast(u16, h); }
__device__ inline float h2f(u16 u)  { return (float)__builtin_bit_cast(_Float16, u); }

union HFrag { v16h h; u32x4 q[2]; };

// A-matrix 16x32 f16 fragment: lane l -> row (l&15), chunks at k0+8*hi and k0+16+8*hi
__device__ inline v16h frag_a(const u16* rowp, int k0, int lane) {
  int hi = lane >> 4;
  const u16* p = rowp + k0 + 8 * hi;
  HFrag f;
  f.q[0] = *(const u32x4*)p;
  f.q[1] = *(const u32x4*)(p + 16);
  return f.h;
}
// B-matrix 32x16 f16 fragment from W stored [N,K] row-major: lane l -> col (l&15),
// chunks at k0+8*hi and k0+16+8*hi
__device__ inline v16h frag_b(const u16* rowp, int k0, int lane) {
  int hi = lane >> 4;
  const u16* p = rowp + k0 + 8 * hi;
  HFrag f;
  f.q[0] = *(const u32x4*)p;
  f.q[1] = *(const u32x4*)(p + 16);
  return f.h;
}

#define WMMA(acc, a, b) \
  acc = __builtin_amdgcn_wmma_f32_16x16x32_f16(false, (a), false, (b), (short)0, (acc), false, false)

// ---------------- pack / convert kernels ----------------
__global__ void pack_pad_kernel(const float* __restrict__ src, u16* __restrict__ dst,
                                int drows, int dcols, int srows, int scols,
                                int s_ld, int s_coloff) {
  int total = drows * dcols;
  for (int i = blockIdx.x * blockDim.x + threadIdx.x; i < total; i += gridDim.x * blockDim.x) {
    int r = i / dcols, c = i - r * dcols;
    float v = (r < srows && c < scols) ? src[(size_t)r * s_ld + s_coloff + c] : 0.f;
    dst[i] = f2h(v);
  }
}

__global__ void pack_x_kernel(const float* __restrict__ obs, const float* __restrict__ reward,
                              u16* __restrict__ x16) {
  int total = kBN * kXK;
  for (int i = blockIdx.x * blockDim.x + threadIdx.x; i < total; i += gridDim.x * blockDim.x) {
    int r = i / kXK, c = i - r * kXK;
    float v = (c < kOBS) ? obs[(size_t)r * kOBS + c] : (c == kOBS ? reward[r] : 0.f);
    x16[i] = f2h(v);
  }
}

__global__ void pack_criw2_kernel(const float* __restrict__ src, u16* __restrict__ dst) {
  int total = kN * 16 * 256;   // per-agent 1x256 row padded to 16x256
  for (int i = blockIdx.x * blockDim.x + threadIdx.x; i < total; i += gridDim.x * blockDim.x) {
    int a = i / (16 * 256);
    int rr = (i / 256) & 15;
    int c = i & 255;
    dst[i] = (rr == 0) ? f2h(src[(size_t)a * 256 + c]) : (u16)0;
  }
}

__global__ void zero_kernel(float* __restrict__ p, int n) {
  for (int i = blockIdx.x * blockDim.x + threadIdx.x; i < n; i += gridDim.x * blockDim.x)
    p[i] = 0.f;
}

// ---------------- GRU: fused gx/gh GEMMs + gates ----------------
__global__ void gru_kernel(const u16* __restrict__ x16, const u16* __restrict__ h016,
                           const u16* __restrict__ wih, const u16* __restrict__ whh,
                           const float* __restrict__ bih, const float* __restrict__ bhh,
                           const float* __restrict__ h0f,
                           float* __restrict__ hout, u16* __restrict__ hid16) {
  int wid = blockIdx.x * (blockDim.x >> 5) + (threadIdx.x >> 5);
  int lane = threadIdx.x & 31;
  constexpr int NT = kRH / 16;            // 16 column tiles
  if (wid >= (kBN / 16) * NT) return;
  int mt = wid / NT;
  int ct = wid - mt * NT;

  const u16* xrow = x16  + (size_t)(mt * 16 + (lane & 15)) * kXK;
  const u16* hrow = h016 + (size_t)(mt * 16 + (lane & 15)) * kRH;
  int wr = ct * 16 + (lane & 15);
  const u16* w_r = wih + (size_t)wr * kXK;
  const u16* w_z = wih + (size_t)(wr + kRH) * kXK;
  const u16* w_n = wih + (size_t)(wr + 2 * kRH) * kXK;
  const u16* u_r = whh + (size_t)wr * kRH;
  const u16* u_z = whh + (size_t)(wr + kRH) * kRH;
  const u16* u_n = whh + (size_t)(wr + 2 * kRH) * kRH;

  v8f axr = {}, axz = {}, axn = {}, ahr = {}, ahz = {}, ahn = {};
  for (int k0 = 0; k0 < kXK; k0 += 32) {
    v16h a = frag_a(xrow, k0, lane);
    WMMA(axr, a, frag_b(w_r, k0, lane));
    WMMA(axz, a, frag_b(w_z, k0, lane));
    WMMA(axn, a, frag_b(w_n, k0, lane));
  }
  for (int k0 = 0; k0 < kRH; k0 += 32) {
    v16h a = frag_a(hrow, k0, lane);
    WMMA(ahr, a, frag_b(u_r, k0, lane));
    WMMA(ahz, a, frag_b(u_z, k0, lane));
    WMMA(ahn, a, frag_b(u_n, k0, lane));
  }
  int hi = lane >> 4;
  int n = ct * 16 + (lane & 15);
  float br = bih[n],            bhr = bhh[n];
  float bz = bih[n + kRH],      bhz = bhh[n + kRH];
  float bn = bih[n + 2 * kRH],  bhn = bhh[n + 2 * kRH];
#pragma unroll
  for (int r = 0; r < 8; r++) {
    int m = mt * 16 + r + 8 * hi;
    float hr = ahr[r] + bhr;
    float hz = ahz[r] + bhz;
    float hn = ahn[r] + bhn;
    float rg = 1.f / (1.f + __expf(-((axr[r] + br) + hr)));
    float zg = 1.f / (1.f + __expf(-((axz[r] + bz) + hz)));
    float nc = tanhf((axn[r] + bn) + rg * hn);
    float h0v = h0f[(size_t)m * kRH + n];
    float h = (1.f - zg) * nc + zg * h0v;
    hout[(size_t)m * kRH + n] = h;
    hid16[(size_t)m * kRH + n] = f2h(h);
  }
}

// ---------------- node projections: hidden[16384,256] x Wcat[1024,256]^T ----------------
__global__ void proj_kernel(const u16* __restrict__ hid16, const u16* __restrict__ wcat,
                            u16* __restrict__ proj16) {
  int wid = blockIdx.x * (blockDim.x >> 5) + (threadIdx.x >> 5);
  int lane = threadIdx.x & 31;
  constexpr int NT = kPROJ / 16;   // 64
  if (wid >= (kBN / 16) * NT) return;
  int mt = wid / NT, nt = wid - mt * NT;
  const u16* arow = hid16 + (size_t)(mt * 16 + (lane & 15)) * kRH;
  const u16* brow = wcat  + (size_t)(nt * 16 + (lane & 15)) * kRH;
  v8f acc = {};
  for (int k0 = 0; k0 < kRH; k0 += 32)
    WMMA(acc, frag_a(arow, k0, lane), frag_b(brow, k0, lane));
  int hi = lane >> 4, n = nt * 16 + (lane & 15);
#pragma unroll
  for (int r = 0; r < 8; r++) {
    int m = mt * 16 + r + 8 * hi;
    proj16[(size_t)m * kPROJ + n] = f2h(acc[r]);
  }
}

// ---------------- msgs: obs[16384,128] x msg_w[1024,128]^T + msg_b ----------------
__global__ void msgs_kernel(const u16* __restrict__ obs16, const u16* __restrict__ msgw,
                            const float* __restrict__ msgb, float* __restrict__ msgs) {
  int wid = blockIdx.x * (blockDim.x >> 5) + (threadIdx.x >> 5);
  int lane = threadIdx.x & 31;
  constexpr int NT = kKOBS / 16;   // 64
  if (wid >= (kBN / 16) * NT) return;
  int mt = wid / NT, nt = wid - mt * NT;
  const u16* arow = obs16 + (size_t)(mt * 16 + (lane & 15)) * kOBS;
  const u16* brow = msgw  + (size_t)(nt * 16 + (lane & 15)) * kOBS;
  v8f acc = {};
  for (int k0 = 0; k0 < kOBS; k0 += 32)
    WMMA(acc, frag_a(arow, k0, lane), frag_b(brow, k0, lane));
  int hi = lane >> 4, n = nt * 16 + (lane & 15);
  float bv = msgb[n];
#pragma unroll
  for (int r = 0; r < 8; r++) {
    int m = mt * 16 + r + 8 * hi;
    msgs[(size_t)m * kKOBS + n] = acc[r] + bv;
  }
}

// ---------------- edge kernel: gather+ReLU -> WMMA second layers -> softmax + scatter ----
__global__ void edge_kernel(const u16* __restrict__ proj16,
                            const float* __restrict__ enc_b1, const float* __restrict__ dec_b1,
                            const u16* __restrict__ encw2, const u16* __restrict__ decw2,
                            const float* __restrict__ enc_b2, const float* __restrict__ dec_b2,
                            float* __restrict__ rel_out, float* __restrict__ scm_out) {
  __shared__ __align__(16) u16 lenc[16 * 256];
  __shared__ __align__(16) u16 ldec[16 * 256];
  __shared__ float llog[16 * 8];
  int b = blockIdx.x / 15;
  int et = blockIdx.x - b * 15;
  int lane = threadIdx.x;

  // phase 1: per-edge pre-activation via node-projection gather
  for (int i = lane; i < 16 * 256; i += 32) {
    int row = i >> 8, c = i & 255;
    int e = et * 16 + row;
    int recv = e / 15;
    int sj = e - recv * 15;
    int send = sj + (sj >= recv);
    const u16* ps = proj16 + (size_t)(b * kN + send) * kPROJ;
    const u16* pr = proj16 + (size_t)(b * kN + recv) * kPROJ;
    float he = h2f(ps[c])       + h2f(pr[256 + c]) + enc_b1[c];
    float hd = h2f(ps[512 + c]) + h2f(pr[768 + c]) + dec_b1[c];
    lenc[i] = f2h(he > 0.f ? he : 0.f);
    ldec[i] = f2h(hd > 0.f ? hd : 0.f);
  }
  __syncthreads();

  // phase 2a: encoder second layer (K->8, padded to 16 cols)
  {
    const u16* arow = lenc + (size_t)(lane & 15) * 256;
    const u16* brow = encw2 + (size_t)(lane & 15) * 256;
    v8f acc = {};
    for (int k0 = 0; k0 < 256; k0 += 32)
      WMMA(acc, frag_a(arow, k0, lane), frag_b(brow, k0, lane));
    int hi = lane >> 4, n = lane & 15;
    if (n < kK) {
#pragma unroll
      for (int r = 0; r < 8; r++) llog[(r + 8 * hi) * kK + n] = acc[r] + enc_b2[n];
    }
  }
  __syncthreads();

  // softmax per edge row -> rel_type
  if (lane < 16) {
    int e = et * 16 + lane;
    float v[kK], mx = -1e30f;
#pragma unroll
    for (int k = 0; k < kK; k++) { v[k] = llog[lane * kK + k]; mx = v[k] > mx ? v[k] : mx; }
    float s = 0.f;
#pragma unroll
    for (int k = 0; k < kK; k++) { v[k] = __expf(v[k] - mx); s += v[k]; }
    float inv = 1.f / s;
#pragma unroll
    for (int k = 0; k < kK; k++)
      rel_out[((size_t)b * kE + e) * kK + k] = v[k] * inv;
  }

  // phase 2b: decoder second layer (256 -> 129, padded to 144) + mean-aggregate scatter
  const float inv_agg = 1.f / (15.f + 1e-6f);
  const u16* arow = ldec + (size_t)(lane & 15) * 256;
  for (int ntile = 0; ntile < kDECN / 16; ntile++) {
    const u16* brow = decw2 + (size_t)(ntile * 16 + (lane & 15)) * 256;
    v8f acc = {};
    for (int k0 = 0; k0 < 256; k0 += 32)
      WMMA(acc, frag_a(arow, k0, lane), frag_b(brow, k0, lane));
    int hi = lane >> 4;
    int col = ntile * 16 + (lane & 15);
    if (col < 129) {
      float bv = dec_b2[col];
#pragma unroll
      for (int r = 0; r < 8; r++) {
        int m = r + 8 * hi;
        int e = et * 16 + m;
        int recv = e / 15;
        atomicAdd(&scm_out[(size_t)(b * kN + recv) * 129 + col], (acc[r] + bv) * inv_agg);
      }
    }
  }
}

// ---------------- message gather + ai staging ----------------
__global__ void message_kernel(const float* __restrict__ rel, const float* __restrict__ msgs,
                               const float* __restrict__ obs, u16* __restrict__ ai16) {
  __shared__ float relw[15 * kK];
  int bi = blockIdx.x;          // 0..BN-1
  int b = bi >> 4, i = bi & 15;
  int t = threadIdx.x;          // 128 threads
  if (t < 15 * kK) {
    int jj = t >> 3, k = t & 7;
    relw[t] = rel[((size_t)b * kE + i * 15 + jj) * kK + k];
  }
  __syncthreads();
  float acc = 0.f;
  for (int jj = 0; jj < 15; jj++) {
    int j = jj + (jj >= i);
    const float* mrow = msgs + (size_t)(b * kN + j) * kKOBS + t;
#pragma unroll
    for (int k = 0; k < kK; k++) acc += relw[jj * kK + k] * mrow[k * kOBS];
  }
  size_t row = (size_t)bi * 256;
  ai16[row + t] = f2h(obs[(size_t)bi * kOBS + t]);
  ai16[row + 128 + t] = f2h(acc);
}

// ---------------- per-agent actor/critic heads ----------------
__global__ void heads_kernel(const u16* __restrict__ ai16,
                             const u16* __restrict__ aw1, const float* __restrict__ ab1,
                             const u16* __restrict__ aw2, const float* __restrict__ ab2,
                             const u16* __restrict__ cw1, const float* __restrict__ cb1,
                             const u16* __restrict__ cw2, const float* __restrict__ cb2,
                             float* __restrict__ actor_out, float* __restrict__ critic_out) {
  __shared__ __align__(16) u16 lhid[16 * 256];
  int agent = blockIdx.x & 15;
  int mt = blockIdx.x >> 4;       // 0..63 batch tiles
  int lane = threadIdx.x;
  const u16* arow = ai16 + ((size_t)(mt * 16 + (lane & 15)) * kN + agent) * 256;

  // actor hidden (256x256) -> ReLU -> LDS
  for (int nt = 0; nt < 16; nt++) {
    const u16* brow = aw1 + ((size_t)agent * 256 + nt * 16 + (lane & 15)) * 256;
    v8f acc = {};
    for (int k0 = 0; k0 < 256; k0 += 32)
      WMMA(acc, frag_a(arow, k0, lane), frag_b(brow, k0, lane));
    int hi = lane >> 4, o = nt * 16 + (lane & 15);
    float bv = ab1[agent * 256 + o];
#pragma unroll
    for (int r = 0; r < 8; r++) {
      float v = acc[r] + bv;
      lhid[(r + 8 * hi) * 256 + o] = f2h(v > 0.f ? v : 0.f);
    }
  }
  __syncthreads();
  // actor output (256 -> 32)
  {
    const u16* hrow = lhid + (size_t)(lane & 15) * 256;
    for (int nt = 0; nt < 2; nt++) {
      const u16* brow = aw2 + ((size_t)agent * kACT + nt * 16 + (lane & 15)) * 256;
      v8f acc = {};
      for (int k0 = 0; k0 < 256; k0 += 32)
        WMMA(acc, frag_a(hrow, k0, lane), frag_b(brow, k0, lane));
      int hi = lane >> 4, col = nt * 16 + (lane & 15);
      float bv = ab2[agent * kACT + col];
#pragma unroll
      for (int r = 0; r < 8; r++) {
        int bcur = mt * 16 + r + 8 * hi;
        actor_out[((size_t)bcur * kN + agent) * kACT + col] = acc[r] + bv;
      }
    }
  }
  __syncthreads();
  // critic hidden (256x256) -> ReLU -> LDS (reuse)
  for (int nt = 0; nt < 16; nt++) {
    const u16* brow = cw1 + ((size_t)agent * 256 + nt * 16 + (lane & 15)) * 256;
    v8f acc = {};
    for (int k0 = 0; k0 < 256; k0 += 32)
      WMMA(acc, frag_a(arow, k0, lane), frag_b(brow, k0, lane));
    int hi = lane >> 4, o = nt * 16 + (lane & 15);
    float bv = cb1[agent * 256 + o];
#pragma unroll
    for (int r = 0; r < 8; r++) {
      float v = acc[r] + bv;
      lhid[(r + 8 * hi) * 256 + o] = f2h(v > 0.f ? v : 0.f);
    }
  }
  __syncthreads();
  // critic output (256 -> 1, row-padded weights to 16)
  {
    const u16* hrow = lhid + (size_t)(lane & 15) * 256;
    const u16* brow = cw2 + ((size_t)agent * 16 + (lane & 15)) * 256;
    v8f acc = {};
    for (int k0 = 0; k0 < 256; k0 += 32)
      WMMA(acc, frag_a(hrow, k0, lane), frag_b(brow, k0, lane));
    int hi = lane >> 4;
    if ((lane & 15) == 0) {
      float bv = cb2[agent];
#pragma unroll
      for (int r = 0; r < 8; r++) {
        int bcur = mt * 16 + r + 8 * hi;
        critic_out[(size_t)bcur * kN + agent] = acc[r] + bv;
      }
    }
  }
}

// ---------------- launch ----------------
static inline int divup(int a, int b) { return (a + b - 1) / b; }

extern "C" void kernel_launch(void* const* d_in, const int* in_sizes, int n_in,
                              void* d_out, int out_size, void* d_ws, size_t ws_size,
                              hipStream_t stream) {
  (void)in_sizes; (void)n_in; (void)out_size; (void)ws_size;
  const float* obs     = (const float*)d_in[0];
  const float* reward  = (const float*)d_in[1];
  const float* h0f     = (const float*)d_in[2];
  const float* gru_wih = (const float*)d_in[3];
  const float* gru_whh = (const float*)d_in[4];
  const float* gru_bih = (const float*)d_in[5];
  const float* gru_bhh = (const float*)d_in[6];
  const float* enc_w1  = (const float*)d_in[7];
  const float* enc_b1  = (const float*)d_in[8];
  const float* enc_w2  = (const float*)d_in[9];
  const float* enc_b2  = (const float*)d_in[10];
  const float* dec_w1  = (const float*)d_in[11];
  const float* dec_b1  = (const float*)d_in[12];
  const float* dec_w2  = (const float*)d_in[13];
  const float* dec_b2  = (const float*)d_in[14];
  const float* msg_w   = (const float*)d_in[15];
  const float* msg_b   = (const float*)d_in[16];
  const float* act_w1  = (const float*)d_in[17];
  const float* act_b1  = (const float*)d_in[18];
  const float* act_w2  = (const float*)d_in[19];
  const float* act_b2  = (const float*)d_in[20];
  const float* cri_w1  = (const float*)d_in[21];
  const float* cri_b1  = (const float*)d_in[22];
  const float* cri_w2  = (const float*)d_in[23];
  const float* cri_b2  = (const float*)d_in[24];

  float* out = (float*)d_out;
  float* actor_out  = out + OFF_ACTOR;
  float* critic_out = out + OFF_CRI;
  float* scm_out    = out + OFF_SCM;
  float* rel_out    = out + OFF_REL;
  float* h_out      = out + OFF_H;

  char* ws = (char*)d_ws;
  size_t off = 0;
  auto alloc = [&](size_t bytes) -> void* {
    void* p = ws + off;
    off = (off + bytes + 255) & ~(size_t)255;
    return p;
  };
  u16* x16     = (u16*)alloc((size_t)kBN * kXK * 2);
  u16* h016    = (u16*)alloc((size_t)kBN * kRH * 2);
  u16* hid16   = (u16*)alloc((size_t)kBN * kRH * 2);
  u16* obs16   = (u16*)alloc((size_t)kBN * kOBS * 2);
  u16* ai16    = (u16*)alloc((size_t)kBN * 256 * 2);
  u16* proj16  = (u16*)alloc((size_t)kBN * kPROJ * 2);
  float* msgsf = (float*)alloc((size_t)kBN * kKOBS * 4);
  u16* wih16   = (u16*)alloc((size_t)3 * kRH * kXK * 2);
  u16* whh16   = (u16*)alloc((size_t)3 * kRH * kRH * 2);
  u16* wcat16  = (u16*)alloc((size_t)kPROJ * kRH * 2);
  u16* encw216 = (u16*)alloc((size_t)16 * 256 * 2);
  u16* decw216 = (u16*)alloc((size_t)kDECN * 256 * 2);
  u16* msgw16  = (u16*)alloc((size_t)kKOBS * kOBS * 2);
  u16* actw116 = (u16*)alloc((size_t)kN * 256 * 256 * 2);
  u16* actw216 = (u16*)alloc((size_t)kN * kACT * 256 * 2);
  u16* criw116 = (u16*)alloc((size_t)kN * 256 * 256 * 2);
  u16* criw216 = (u16*)alloc((size_t)kN * 16 * 256 * 2);

  const int TB = 256;
  // activation packs
  pack_x_kernel<<<divup(kBN * kXK, TB), TB, 0, stream>>>(obs, reward, x16);
  pack_pad_kernel<<<divup(kBN * kRH, TB), TB, 0, stream>>>(h0f, h016, kBN, kRH, kBN, kRH, kRH, 0);
  pack_pad_kernel<<<divup(kBN * kOBS, TB), TB, 0, stream>>>(obs, obs16, kBN, kOBS, kBN, kOBS, kOBS, 0);
  // weight packs (all stored [out,in] row-major, K padded/zero-filled)
  pack_pad_kernel<<<divup(3 * kRH * kXK, TB), TB, 0, stream>>>(gru_wih, wih16, 3 * kRH, kXK, 3 * kRH, 129, 129, 0);
  pack_pad_kernel<<<divup(3 * kRH * kRH, TB), TB, 0, stream>>>(gru_whh, whh16, 3 * kRH, kRH, 3 * kRH, kRH, kRH, 0);
  // Wcat rows: [enc_w1a | enc_w1b | dec_w1a | dec_w1b], each 256x256
  pack_pad_kernel<<<divup(256 * 256, TB), TB, 0, stream>>>(enc_w1, wcat16 + 0 * 256 * 256, 256, 256, 256, 256, 512, 0);
  pack_pad_kernel<<<divup(256 * 256, TB), TB, 0, stream>>>(enc_w1, wcat16 + 1 * 256 * 256, 256, 256, 256, 256, 512, 256);
  pack_pad_kernel<<<divup(256 * 256, TB), TB, 0, stream>>>(dec_w1, wcat16 + 2 * 256 * 256, 256, 256, 256, 256, 512, 0);
  pack_pad_kernel<<<divup(256 * 256, TB), TB, 0, stream>>>(dec_w1, wcat16 + 3 * 256 * 256, 256, 256, 256, 256, 512, 256);
  pack_pad_kernel<<<divup(16 * 256, TB), TB, 0, stream>>>(enc_w2, encw216, 16, 256, kK, 256, 256, 0);
  pack_pad_kernel<<<divup(kDECN * 256, TB), TB, 0, stream>>>(dec_w2, decw216, kDECN, 256, 129, 256, 256, 0);
  pack_pad_kernel<<<divup(kKOBS * kOBS, TB), TB, 0, stream>>>(msg_w, msgw16, kKOBS, kOBS, kKOBS, kOBS, kOBS, 0);
  pack_pad_kernel<<<divup(kN * 256 * 256, TB), TB, 0, stream>>>(act_w1, actw116, kN * 256, 256, kN * 256, 256, 256, 0);
  pack_pad_kernel<<<divup(kN * kACT * 256, TB), TB, 0, stream>>>(act_w2, actw216, kN * kACT, 256, kN * kACT, 256, 256, 0);
  pack_pad_kernel<<<divup(kN * 256 * 256, TB), TB, 0, stream>>>(cri_w1, criw116, kN * 256, 256, kN * 256, 256, 256, 0);
  pack_criw2_kernel<<<divup(kN * 16 * 256, TB), TB, 0, stream>>>(cri_w2, criw216);
  // scm output accumulated via atomics -> zero first
  zero_kernel<<<divup(SCM_ELEMS, TB), TB, 0, stream>>>(scm_out, SCM_ELEMS);

  // GRU: (BN/16)*(RH/16) = 16384 waves, 4 waves/block
  gru_kernel<<<(kBN / 16) * (kRH / 16) / 4, 128, 0, stream>>>(
      x16, h016, wih16, whh16, gru_bih, gru_bhh, h0f, h_out, hid16);
  // node projections: 65536 waves
  proj_kernel<<<(kBN / 16) * (kPROJ / 16) / 4, 128, 0, stream>>>(hid16, wcat16, proj16);
  // per-type message linears: 65536 waves
  msgs_kernel<<<(kBN / 16) * (kKOBS / 16) / 4, 128, 0, stream>>>(obs16, msgw16, msg_b, msgsf);
  // edges: B * (E/16) blocks, one wave each
  edge_kernel<<<kB * (kE / 16), 32, 0, stream>>>(
      proj16, enc_b1, dec_b1, encw216, decw216, enc_b2, dec_b2, rel_out, scm_out);
  // message gather + ai staging
  message_kernel<<<kBN, 128, 0, stream>>>(rel_out, msgsf, obs, ai16);
  // per-agent heads: 16 agents * 64 batch tiles
  heads_kernel<<<kN * (kB / 16), 32, 0, stream>>>(
      ai16, actw116, act_b1, actw216, act_b2, criw116, cri_b1, criw216, cri_b2,
      actor_out, critic_out);
}
